// SimilarAttentionConv_56023553409779
// MI455X (gfx1250) — compile-verified, matrix operand order checked by reading, not run on hardware
//
#include <hip/hip_runtime.h>

// ---------------------------------------------------------------------------
// AGNN (SimilarAttentionConv) for MI455X / gfx1250, wave32, WMMA bf16.
// N=8192 nodes, IN=256, HID=128, E=262144 edges (+N self loops).
// Dominant cost: out = adj @ h  -> bandwidth-bound on adj.
// adj is read exactly once (wave computes full 16x128 strip), the big mm is
// K-split 8-way for occupancy, and A is streamed with non-temporal loads.
// ---------------------------------------------------------------------------

#define HIDC   128   // hidden channels (fixed by the model)
#define NTILES 8     // HIDC / 16 output column tiles per wave

typedef __attribute__((ext_vector_type(16))) __bf16 v16bf;
typedef __attribute__((ext_vector_type(8)))  float  v8f;
typedef __attribute__((ext_vector_type(4)))  float  nf4;   // native vec for NT loads

union BfFrag {
    unsigned short us[16];
    uint4          q[2];
    v16bf          v;
};

__device__ __forceinline__ unsigned short f2bf(float f) {
    // round-to-nearest-even f32 -> bf16 (bit-level; exact for 0.0/1.0 of adj)
    unsigned u = __float_as_uint(f);
    unsigned r = u + 0x7FFFu + ((u >> 16) & 1u);
    return (unsigned short)(r >> 16);
}

// ---------------------------------------------------------------------------
// small utility kernels
// ---------------------------------------------------------------------------
__global__ void k_zero_f32(float* __restrict__ p, int n) {
    int i = blockIdx.x * 256 + threadIdx.x;
    if (i < n) p[i] = 0.0f;
}

__global__ void k_cvt_bf16(const float* __restrict__ s, unsigned short* __restrict__ d, int n) {
    int i = blockIdx.x * 256 + threadIdx.x;
    if (i < n) d[i] = f2bf(s[i]);
}

// ---------------------------------------------------------------------------
// WMMA GEMM:  C[M x 128] = A[M x K] (f32) * Bt[128 x K] (bf16)^T (+ bias)
//
// One wave per (16-row strip, K-chunk). The wave holds 8 accumulators and
// computes the full 16x128 strip, so each A fragment (adj!) is loaded ONCE
// and reused against all 8 B tiles. A is streamed with non-temporal loads.
//
// Fragment layout per CDNA5 ISA 7.12.2 (wave32):
//   lane L: half = L>>4, r = L&15, kb = k + half*8
//   A frag: row M=r, us[0..7]=K[kb..kb+7], us[8..15]=K[kb+16..kb+23]
//   B frag: col N=r, same K pattern (Bt rows contiguous in K -> 16B loads)
//   C/D:   acc[j] = C[strip*16 + j + 8*half][t*16 + r]
//
// accumulate=0: epilogue adds bias, optional relu, stores f32 and/or
//               transposed-bf16 (the Bt operand layout of a following GEMM).
// accumulate=1: K-split partials combined via f32 atomicAdd (outF pre-zeroed).
// ---------------------------------------------------------------------------
__global__ __launch_bounds__(256)
void k_gemm_wmma(const float* __restrict__ A,
                 const unsigned short* __restrict__ Bt,
                 const float* __restrict__ bias,
                 float* __restrict__ outF,
                 unsigned short* __restrict__ outT,
                 int M, int K, int relu, int kChunk, int accumulate)
{
    int wave  = (blockIdx.x * blockDim.x + threadIdx.x) >> 5;
    int lane  = threadIdx.x & 31;
    int nKc   = K / kChunk;                    // K-chunks per strip
    int strip = wave / nKc;
    int kc    = wave - strip * nKc;
    if (strip * 16 >= M) return;               // wave-uniform guard

    int half = lane >> 4;
    int r15  = lane & 15;
    int k0   = kc * kChunk;
    int k1   = k0 + kChunk;

    const float*          ap = A  + (long)(strip * 16 + r15) * K;
    const unsigned short* bp = Bt + (long)r15 * K;            // + t*16*K per tile

    v8f acc[NTILES];
#pragma unroll
    for (int t = 0; t < NTILES; ++t) acc[t] = (v8f){};

    BfFrag fa, fb;
    const long bStride = (long)16 * K;                        // Bt rows per tile

    for (int k = k0; k < k1; k += 32) {
        int kb = k + half * 8;
        // A: 2 x 32B contiguous f32 loads (non-temporal: zero-reuse stream),
        //    RNE-convert to bf16 in-register (exact for adj's {0,1}).
        nf4 a0 = __builtin_nontemporal_load((const nf4*)(ap + kb));
        nf4 a1 = __builtin_nontemporal_load((const nf4*)(ap + kb + 4));
        nf4 a2 = __builtin_nontemporal_load((const nf4*)(ap + kb + 16));
        nf4 a3 = __builtin_nontemporal_load((const nf4*)(ap + kb + 20));
#pragma unroll
        for (int i = 0; i < 4; ++i) {
            fa.us[i]      = f2bf(a0[i]);
            fa.us[4 + i]  = f2bf(a1[i]);
            fa.us[8 + i]  = f2bf(a2[i]);
            fa.us[12 + i] = f2bf(a3[i]);
        }

        const unsigned short* bk = bp + kb;
#pragma unroll
        for (int t = 0; t < NTILES; ++t) {
            // B: 2 x 16B contiguous bf16 loads (hot operand, L2-resident)
            fb.q[0] = *(const uint4*)(bk);
            fb.q[1] = *(const uint4*)(bk + 16);
            bk += bStride;
            acc[t] = __builtin_amdgcn_wmma_f32_16x16x32_bf16(
                /*neg_a=*/false, fa.v, /*neg_b=*/false, fb.v,
                /*c_mod=*/(short)0, acc[t], /*reuse_a=*/false, /*reuse_b=*/false);
        }
    }

    if (accumulate) {
#pragma unroll
        for (int t = 0; t < NTILES; ++t) {
            int col = t * 16 + r15;
#pragma unroll
            for (int j = 0; j < 8; ++j) {
                int row = strip * 16 + j + half * 8;
                atomicAdd(&outF[(long)row * HIDC + col], acc[t][j]);
            }
        }
    } else {
#pragma unroll
        for (int t = 0; t < NTILES; ++t) {
            int   col = t * 16 + r15;
            float bv  = bias ? bias[col] : 0.0f;
#pragma unroll
            for (int j = 0; j < 8; ++j) {
                int   row = strip * 16 + j + half * 8;
                float v   = acc[t][j] + bv;
                if (relu) v = fmaxf(v, 0.0f);
                if (outF) outF[(long)row * HIDC + col] = v;
                if (outT) outT[(long)col * M + row] = f2bf(v);
            }
        }
    }
}

// ---------------------------------------------------------------------------
// AGNN propagation pieces (h is [N x 128] f32, L2-resident at 4 MB)
// ---------------------------------------------------------------------------

// L2 norm per node: one wave per node, float4 per lane, shfl-xor reduce.
__global__ void k_norm(const float* __restrict__ h, float* __restrict__ nrm, int Nn)
{
    int node = (blockIdx.x * blockDim.x + threadIdx.x) >> 5;
    int lane = threadIdx.x & 31;
    if (node >= Nn) return;
    float4 v = *(const float4*)(h + (long)node * HIDC + lane * 4);
    float  s = v.x * v.x + v.y * v.y + v.z * v.z + v.w * v.w;
#pragma unroll
    for (int o = 16; o > 0; o >>= 1) s += __shfl_xor(s, o, 32);
    if (lane == 0) nrm[node] = fmaxf(sqrtf(s), 1e-12f);
}

// Per-edge cosine score + segment-max via monotone-encoded u32 atomicMax.
__global__ void k_alpha(const float* __restrict__ h, const float* __restrict__ nrm,
                        const int* __restrict__ src, const int* __restrict__ dst,
                        int E, int Etot, const float* __restrict__ betap,
                        float* __restrict__ alpha, unsigned* __restrict__ menc)
{
    int e    = (blockIdx.x * blockDim.x + threadIdx.x) >> 5;
    int lane = threadIdx.x & 31;
    if (e >= Etot) return;
    int s, d;
    if (e < E) { s = src[e]; d = dst[e]; } else { s = d = e - E; }  // self loop
    float4 a = *(const float4*)(h + (long)s * HIDC + lane * 4);
    float4 b = *(const float4*)(h + (long)d * HIDC + lane * 4);
    float dot = a.x * b.x + a.y * b.y + a.z * b.z + a.w * b.w;
#pragma unroll
    for (int o = 16; o > 0; o >>= 1) dot += __shfl_xor(dot, o, 32);
    if (lane == 0) {
        float beta = betap ? betap[0] : 1.0f;
        float al   = beta * dot / (nrm[s] * nrm[d]);
        alpha[e]   = al;
        unsigned u   = __float_as_uint(al);
        unsigned enc = (u & 0x80000000u) ? ~u : (u | 0x80000000u);  // order-preserving
        atomicMax(&menc[d], enc);
    }
}

// exp(alpha - max) and segment-sum via f32 atomicAdd. One thread per edge.
__global__ void k_expsum(const float* __restrict__ alpha, const int* __restrict__ dst,
                         const unsigned* __restrict__ menc,
                         float* __restrict__ ew, float* __restrict__ ssum, int E, int Etot)
{
    int e = blockIdx.x * 256 + threadIdx.x;
    if (e >= Etot) return;
    int d = (e < E) ? dst[e] : (e - E);
    unsigned enc = menc[d];
    unsigned u   = (enc & 0x80000000u) ? (enc ^ 0x80000000u) : ~enc;
    float    m   = __uint_as_float(u);
    float    v   = expf(alpha[e] - m);
    ew[e] = v;
    atomicAdd(&ssum[d], v);
}

// Weighted scatter-add of source features. One wave per edge, 4 chan/lane.
__global__ void k_scatter(const float* __restrict__ h, const float* __restrict__ ew,
                          const float* __restrict__ ssum,
                          const int* __restrict__ src, const int* __restrict__ dst,
                          float* __restrict__ hout, int E, int Etot)
{
    int e    = (blockIdx.x * blockDim.x + threadIdx.x) >> 5;
    int lane = threadIdx.x & 31;
    if (e >= Etot) return;
    int s, d;
    if (e < E) { s = src[e]; d = dst[e]; } else { s = d = e - E; }
    float  w = ew[e] / (ssum[d] + 1e-16f);
    float4 a = *(const float4*)(h + (long)s * HIDC + lane * 4);
    float* o = hout + (long)d * HIDC + lane * 4;
    atomicAdd(o + 0, a.x * w);
    atomicAdd(o + 1, a.y * w);
    atomicAdd(o + 2, a.z * w);
    atomicAdd(o + 3, a.w * w);
}

// ---------------------------------------------------------------------------
// host-side orchestration
// ---------------------------------------------------------------------------
static inline size_t alignUp(size_t v, size_t a) { return (v + a - 1) & ~(a - 1); }

extern "C" void kernel_launch(void* const* d_in, const int* in_sizes, int n_in,
                              void* d_out, int out_size, void* d_ws, size_t ws_size,
                              hipStream_t stream)
{
    const float* x     = (const float*)d_in[0];
    const float* adj   = (const float*)d_in[1];
    const int*   ei    = (const int*)  d_in[2];
    const float* W1    = (const float*)d_in[3];
    const float* b1    = (const float*)d_in[4];
    const float* W2    = (const float*)d_in[5];
    const float* b2    = (const float*)d_in[6];
    const float* beta2 = (const float*)d_in[7];

    const int E    = in_sizes[2] / 2;         // edge_index is [2, E]
    const int HID  = in_sizes[4];             // 128
    const int IN   = in_sizes[3] / HID;       // 256
    const int Nn   = in_sizes[0] / IN;        // 8192
    const int Etot = E + Nn;                  // with self loops

    const int* srcA = ei;
    const int* dstA = ei + E;

    // ---- workspace carve-up (all 256B aligned) ----
    char*  base = (char*)d_ws;
    size_t off  = 0;
    auto carve = [&](size_t bytes) {
        char* p = base + off;
        off = alignUp(off + bytes, 256);
        return p;
    };
    float*          h1    = (float*)         carve((size_t)Nn * HID * 4);
    float*          h2    = (float*)         carve((size_t)Nn * HID * 4);
    float*          h3    = (float*)         carve((size_t)Nn * HID * 4);
    float*          nrm   = (float*)         carve((size_t)Nn * 4);
    unsigned*       menc  = (unsigned*)      carve((size_t)Nn * 4);
    float*          ssum  = (float*)         carve((size_t)Nn * 4);
    float*          alpha = (float*)         carve((size_t)Etot * 4);
    float*          ew    = (float*)         carve((size_t)Etot * 4);
    unsigned short* W1b   = (unsigned short*)carve((size_t)HID * IN * 2);
    unsigned short* W2b   = (unsigned short*)carve((size_t)HID * HID * 2);
    unsigned short* h4t   = (unsigned short*)carve((size_t)HID * Nn * 2);
    (void)ws_size;

    const int TB = 256;
    auto blocks1 = [&](int n)     { return (n + TB - 1) / TB; };          // 1 thread/item
    auto blocksW = [&](int waves) { return (waves * 32 + TB - 1) / TB; }; // 1 wave/item

    // weights -> bf16 (Bt layout: W[n][k] is already [Nc x K])
    k_cvt_bf16<<<blocks1(HID * IN),  TB, 0, stream>>>(W1, W1b, HID * IN);
    k_cvt_bf16<<<blocks1(HID * HID), TB, 0, stream>>>(W2, W2b, HID * HID);

    // h1 = relu(x @ W1^T + b1): 512 strip-waves, full K
    k_gemm_wmma<<<blocksW(Nn / 16), TB, 0, stream>>>(
        x, W1b, b1, h1, nullptr, Nn, IN, /*relu=*/1, /*kChunk=*/IN, /*acc=*/0);

    // AGNN propagation (twice)
    auto prop = [&](const float* hin, float* hout, const float* betap) {
        k_zero_f32<<<blocks1(Nn),       TB, 0, stream>>>((float*)menc, Nn);
        k_zero_f32<<<blocks1(Nn),       TB, 0, stream>>>(ssum, Nn);
        k_zero_f32<<<blocks1(Nn * HID), TB, 0, stream>>>(hout, Nn * HID);
        k_norm   <<<blocksW(Nn),        TB, 0, stream>>>(hin, nrm, Nn);
        k_alpha  <<<blocksW(Etot),      TB, 0, stream>>>(hin, nrm, srcA, dstA,
                                                         E, Etot, betap, alpha, menc);
        k_expsum <<<blocks1(Etot),      TB, 0, stream>>>(alpha, dstA, menc, ew, ssum, E, Etot);
        k_scatter<<<blocksW(Etot),      TB, 0, stream>>>(hin, ew, ssum, srcA, dstA,
                                                         hout, E, Etot);
    };
    prop(h1, h2, nullptr);   // prop1: beta fixed at 1.0
    prop(h2, h3, beta2);     // prop2: learnable beta

    // h4 = relu(h3 @ W2^T + b2), stored transposed in bf16 -> Bt of final mm
    k_gemm_wmma<<<blocksW(Nn / 16), TB, 0, stream>>>(
        h3, W2b, b2, nullptr, h4t, Nn, HID, /*relu=*/1, /*kChunk=*/HID, /*acc=*/0);

    // out = adj @ h4: adj (256 MB) read exactly once, NT-streamed.
    // K split 8-way (kChunk=1024) -> 4096 waves; partials via f32 atomicAdd.
    k_zero_f32<<<blocks1(Nn * HID), TB, 0, stream>>>((float*)d_out, Nn * HID);
    k_gemm_wmma<<<blocksW((Nn / 16) * 8), TB, 0, stream>>>(
        adj, h4t, nullptr, (float*)d_out, nullptr, Nn, Nn,
        /*relu=*/0, /*kChunk=*/Nn / 8, /*acc=*/1);
}
